// QuantBasicTransformerBlock_11029476016405
// MI455X (gfx1250) — matrix operand order repaired and checked by reading, not compiled, not run on hardware
//
#include <hip/hip_runtime.h>

// ---------------------------------------------------------------------------
// Probe results (round 2): toolchain HAS global/cluster async-to-LDS builtins,
// s_wait_asynccnt, tensor_load_to_lds, s_wait_tensorcnt. MISSING ds_load_tr*,
// global_load_tr*. Async builtin: (as1 v4i*, as3 v4i*, imm offset, cpol).
// ---------------------------------------------------------------------------
#if __has_builtin(__builtin_amdgcn_global_load_async_to_lds_b128) && \
    __has_builtin(__builtin_amdgcn_s_wait_asynccnt)
#define HAVE_ASYNC_LDS 1
#else
#define HAVE_ASYNC_LDS 0
#endif

typedef int v4i __attribute__((vector_size(16)));
#if HAVE_ASYNC_LDS
typedef __attribute__((address_space(1))) v4i* gas_v4i_ptr;
typedef __attribute__((address_space(3))) v4i* las_v4i_ptr;
#endif

// ---------------------------------------------------------------------------
// Types / WMMA helpers (CDNA5 gfx1250, wave32)
// ---------------------------------------------------------------------------
typedef __attribute__((ext_vector_type(16))) _Float16 v16h;
typedef __attribute__((ext_vector_type(8)))  _Float16 v8h;
typedef __attribute__((ext_vector_type(8)))  float    v8f;

static __device__ __forceinline__ v8f wmma_f32_16x16x32(v16h a, v16h b, v8f c) {
  return __builtin_amdgcn_wmma_f32_16x16x32_f16(false, a, false, b, (short)0, c,
                                                false, false);
}

static __device__ __forceinline__ v16h cat8(v8h lo, v8h hi) {
  return __builtin_shufflevector(lo, hi, 0, 1, 2, 3, 4, 5, 6, 7,
                                 8, 9, 10, 11, 12, 13, 14, 15);
}
// A-matrix 16x32 f16 fragment: per lane-half, contiguous 16 K values [p..p+15]
static __device__ __forceinline__ v16h load_fragA(const _Float16* p) {
  return cat8(*(const v8h*)p, *(const v8h*)(p + 8));
}
// B-matrix 32x16 f16 fragment: per lane-half, contiguous 16 K values.
static __device__ __forceinline__ v16h load_fragB(const _Float16* p) {
  return cat8(*(const v8h*)p, *(const v8h*)(p + 8));
}

static __device__ __forceinline__ float fake_quant(float x, float delta) {
  float q = rintf(x / delta) + 128.f;
  q = fminf(fmaxf(q, 0.f), 255.f);
  return (q - 128.f) * delta;
}

static __device__ __forceinline__ void wait_async_then_barrier() {
#if HAVE_ASYNC_LDS
  __builtin_amdgcn_s_wait_asynccnt(0);
#endif
  __syncthreads();
}

#define HEADS 16
#define DHEAD 64

// ---------------------------------------------------------------------------
// fp32 -> f16 cast (weights / context)
// ---------------------------------------------------------------------------
__global__ void cast_f16_kernel(const float* __restrict__ in,
                                _Float16* __restrict__ out, long n) {
  long i = (long)blockIdx.x * 256 + threadIdx.x;
  long st = (long)gridDim.x * 256;
  for (; i < n; i += st) out[i] = (_Float16)in[i];
}

// ---------------------------------------------------------------------------
// LayerNorm (one block per row) -> f16
// ---------------------------------------------------------------------------
__global__ __launch_bounds__(256) void ln_f16_kernel(
    const float* __restrict__ x, const float* __restrict__ w,
    const float* __restrict__ b, _Float16* __restrict__ out, int D) {
  __shared__ float red[256];
  const int row = blockIdx.x;
  const float* xr = x + (size_t)row * D;
  float s = 0.f;
  for (int i = threadIdx.x; i < D; i += 256) s += xr[i];
  red[threadIdx.x] = s;
  __syncthreads();
  for (int st = 128; st > 0; st >>= 1) {
    if (threadIdx.x < st) red[threadIdx.x] += red[threadIdx.x + st];
    __syncthreads();
  }
  float mean = red[0] / (float)D;
  __syncthreads();
  float s2 = 0.f;
  for (int i = threadIdx.x; i < D; i += 256) {
    float d = xr[i] - mean;
    s2 += d * d;
  }
  red[threadIdx.x] = s2;
  __syncthreads();
  for (int st = 128; st > 0; st >>= 1) {
    if (threadIdx.x < st) red[threadIdx.x] += red[threadIdx.x + st];
    __syncthreads();
  }
  float rstd = rsqrtf(red[0] / (float)D + 1e-5f);
  for (int i = threadIdx.x; i < D; i += 256)
    out[(size_t)row * D + i] = (_Float16)((xr[i] - mean) * rstd * w[i] + b[i]);
}

// ---------------------------------------------------------------------------
// Tiled f16 WMMA GEMM: C[M,N] = A[M,K] * B[K,N] (+bias)(+res), f32 out.
// Block = 256 threads (8 waves), tile 128x128, BK=32, double-buffered LDS.
// Wave (wr,wc) owns a 32x64 quadrant -> 2x4 accumulators of v8f.
// Pipeline: issue next tile's async-A (ASYNCcnt) + global-B loads, compute
// current tile's 8 WMMAs while in flight, scatter B, single barrier.
// ---------------------------------------------------------------------------
#define BM 128
#define BN 128
#define BK 32
__global__ __launch_bounds__(256) void gemm_f16_kernel(
    const _Float16* __restrict__ A, const _Float16* __restrict__ B,
    float* __restrict__ C, const float* __restrict__ bias,
    const float* __restrict__ res, int M, int N, int K) {
  __shared__ __attribute__((aligned(16))) _Float16 As[2][BM][BK];
  __shared__ __attribute__((aligned(16))) _Float16 BsT[2][BN][BK];  // [n][k]

  const int tid = threadIdx.x;
  const int lane = tid & 31, wave = tid >> 5;
  const int wr = wave & 3, wc = wave >> 2;
  const int lmod = lane & 15, lhalf = lane >> 4;
  const int tM = blockIdx.y * BM, tN = blockIdx.x * BN;

  // loop-invariant staging assignments
  const int arow = tid >> 1;             // 0..127
  const int aseg = (tid & 1) * 16;       // 0 / 16
  const bool aval = (tM + arow) < M;
  const int bk = tid >> 3;               // 0..31
  const int bns = (tid & 7) * 16;        // 0..112
  const _Float16* ag = A + (size_t)(tM + arow) * K + aseg;  // advances by BK
  const _Float16* bg = B + (size_t)bk * N + tN + bns;       // advances by BK*N

  v8f acc[2][4] = {};

  // ---- prologue: stage tile 0 into buffer 0
#if HAVE_ASYNC_LDS
  if (aval) {
    __builtin_amdgcn_global_load_async_to_lds_b128(
        (gas_v4i_ptr)ag, (las_v4i_ptr)&As[0][arow][aseg], 0, 0);
    __builtin_amdgcn_global_load_async_to_lds_b128(
        (gas_v4i_ptr)ag, (las_v4i_ptr)&As[0][arow][aseg], 16, 0);
  } else {
    v8h z = {};
    *(v8h*)&As[0][arow][aseg] = z;
    *(v8h*)&As[0][arow][aseg + 8] = z;
  }
#else
  {
    v8h a0 = {}, a1 = {};
    if (aval) { a0 = *(const v8h*)ag; a1 = *(const v8h*)(ag + 8); }
    *(v8h*)&As[0][arow][aseg] = a0;
    *(v8h*)&As[0][arow][aseg + 8] = a1;
  }
#endif
  {
    v8h t0 = *(const v8h*)bg;
    v8h t1 = *(const v8h*)(bg + 8);
#pragma unroll
    for (int i = 0; i < 8; i++) BsT[0][bns + i][bk] = t0[i];
#pragma unroll
    for (int i = 0; i < 8; i++) BsT[0][bns + 8 + i][bk] = t1[i];
  }
  wait_async_then_barrier();

  const _Float16* agn = ag + BK;
  const _Float16* bgn = bg + (size_t)BK * N;
  int buf = 0;

  for (int k0 = 0; k0 < K; k0 += BK) {
    const bool hn = (k0 + BK) < K;
    v8h nb0 = {}, nb1 = {};
    if (hn) {
      // ---- issue next tile's loads (A via async engine, B via loadcnt)
#if HAVE_ASYNC_LDS
      if (aval) {
        __builtin_amdgcn_global_load_async_to_lds_b128(
            (gas_v4i_ptr)agn, (las_v4i_ptr)&As[buf ^ 1][arow][aseg], 0, 0);
        __builtin_amdgcn_global_load_async_to_lds_b128(
            (gas_v4i_ptr)agn, (las_v4i_ptr)&As[buf ^ 1][arow][aseg], 16, 0);
      } else {
        v8h z = {};
        *(v8h*)&As[buf ^ 1][arow][aseg] = z;
        *(v8h*)&As[buf ^ 1][arow][aseg + 8] = z;
      }
#else
      if (aval) {
        v8h a0 = *(const v8h*)agn;
        v8h a1 = *(const v8h*)(agn + 8);
        *(v8h*)&As[buf ^ 1][arow][aseg] = a0;
        *(v8h*)&As[buf ^ 1][arow][aseg + 8] = a1;
      } else {
        v8h z = {};
        *(v8h*)&As[buf ^ 1][arow][aseg] = z;
        *(v8h*)&As[buf ^ 1][arow][aseg + 8] = z;
      }
#endif
      nb0 = *(const v8h*)bgn;
      nb1 = *(const v8h*)(bgn + 8);
      if (k0 + 2 * BK < K) {  // warm L2 two tiles ahead
        __builtin_prefetch(agn + BK, 0, 0);
        __builtin_prefetch(bgn + (size_t)BK * N, 0, 0);
      }
    }

    // ---- compute current tile: 8 WMMAs while next-tile loads are in flight
    v16h af0 = load_fragA(&As[buf][wr * 32 + lmod][lhalf * 16]);
    v16h af1 = load_fragA(&As[buf][wr * 32 + 16 + lmod][lhalf * 16]);
#pragma unroll
    for (int c = 0; c < 4; c++) {
      v16h bf = load_fragB(&BsT[buf][wc * 64 + c * 16 + lmod][lhalf * 16]);
      acc[0][c] = wmma_f32_16x16x32(af0, bf, acc[0][c]);
      acc[1][c] = wmma_f32_16x16x32(af1, bf, acc[1][c]);
    }

    if (hn) {
      // ---- scatter next B tile into the other buffer (transpose in LDS)
#pragma unroll
      for (int i = 0; i < 8; i++) BsT[buf ^ 1][bns + i][bk] = nb0[i];
#pragma unroll
      for (int i = 0; i < 8; i++) BsT[buf ^ 1][bns + 8 + i][bk] = nb1[i];
      agn += BK;
      bgn += (size_t)BK * N;
      wait_async_then_barrier();
      buf ^= 1;
    }
  }

  // ---- epilogue: row = base + lhalf*8 + e, col = base + lmod
#pragma unroll
  for (int r = 0; r < 2; r++) {
#pragma unroll
    for (int c = 0; c < 4; c++) {
      int row0 = tM + wr * 32 + r * 16 + lhalf * 8;
      int col = tN + wc * 64 + c * 16 + lmod;
      float bv = bias ? bias[col] : 0.f;
#pragma unroll
      for (int e = 0; e < 8; e++) {
        int row = row0 + e;
        if (row < M) {
          float v = acc[r][c][e] + bv;
          if (res) v += res[(size_t)row * N + col];
          C[(size_t)row * N + col] = v;
        }
      }
    }
  }
}

// ---------------------------------------------------------------------------
// Fake-quant + head split: src f32 [B*rows, 1024] -> dst f16 [B,H,npad,64]
// ---------------------------------------------------------------------------
__global__ void quant_heads_kernel(const float* __restrict__ src,
                                   _Float16* __restrict__ dst,
                                   const float* __restrict__ dp, int rows,
                                   int npad) {
  const float delta = *dp;
  int n = blockIdx.x, h = blockIdx.y, b = blockIdx.z, d = threadIdx.x;
  float v = 0.f;
  if (n < rows)
    v = fake_quant(src[((size_t)b * rows + n) * (HEADS * DHEAD) + h * DHEAD + d],
                   delta);
  dst[(((size_t)b * HEADS + h) * npad + n) * DHEAD + d] = (_Float16)v;
}

// transposed variant for V: dst f16 [B,H,64,npad]
__global__ void quant_headsT_kernel(const float* __restrict__ src,
                                    _Float16* __restrict__ dst,
                                    const float* __restrict__ dp, int rows,
                                    int npad) {
  const float delta = *dp;
  int n = blockIdx.x, h = blockIdx.y, b = blockIdx.z, d = threadIdx.x;
  float v = 0.f;
  if (n < rows)
    v = fake_quant(src[((size_t)b * rows + n) * (HEADS * DHEAD) + h * DHEAD + d],
                   delta);
  dst[(((size_t)b * HEADS + h) * DHEAD + d) * npad + n] = (_Float16)v;
}

// ---------------------------------------------------------------------------
// Fused attention: one block per (b, h, 16-query tile). 128 threads = 4 waves.
//  phase 1: S = Q K^T * scale via WMMA -> LDS score strip [16][NKVPAD] (f32)
//  phase 2: exact row max / sum(exp) (post-softmax fake-quant stays exact)
//  phase 3: P quantized on the fly, O = P V via WMMA, store f16 [B,NQ,1024]
// ---------------------------------------------------------------------------
__global__ __launch_bounds__(128) void attn_kernel(
    const _Float16* __restrict__ Q,   // [B,H,NQ,64]
    const _Float16* __restrict__ Kk,  // [B,H,NKVPAD,64]
    const _Float16* __restrict__ Vt,  // [B,H,64,NKVPAD]
    _Float16* __restrict__ O,         // [B,NQ,H*64]
    const float* __restrict__ dwp, int NQ, int NKV, int NKVPAD, float scale) {
  extern __shared__ float sc[];  // [16][NKVPAD]
  __shared__ float partial[128];
  __shared__ float rowmax[16];
  __shared__ float rowinv[16];

  const int tid = threadIdx.x;
  const int lane = tid & 31, wave = tid >> 5;
  const int lmod = lane & 15, lhalf = lane >> 4;
  const int qt = blockIdx.x, h = blockIdx.y, b = blockIdx.z;
  const int Hh = gridDim.y;

  const float dw = *dwp;
  const float rdw = 1.f / dw;

  const _Float16* qrow =
      Q + (((size_t)b * Hh + h) * NQ + (size_t)qt * 16 + lmod) * DHEAD;
  v16h aq[2];
#pragma unroll
  for (int t = 0; t < 2; t++)
    aq[t] = load_fragA(qrow + t * 32 + lhalf * 16);

  // ---- phase 1: scores
  const int nblk = NKVPAD >> 4;
  for (int jb = wave; jb < nblk; jb += 4) {
    const _Float16* kp =
        Kk + (((size_t)b * Hh + h) * NKVPAD + (size_t)jb * 16 + lmod) * DHEAD;
    v16h bk0 = load_fragB(kp + 0 * 32 + lhalf * 16);
    v16h bk1 = load_fragB(kp + 1 * 32 + lhalf * 16);
    v8f s = {};
    s = wmma_f32_16x16x32(aq[0], bk0, s);
    s = wmma_f32_16x16x32(aq[1], bk1, s);
#pragma unroll
    for (int e = 0; e < 8; e++)
      sc[(size_t)(lhalf * 8 + e) * NKVPAD + jb * 16 + lmod] = s[e] * scale;
  }
  __syncthreads();

  // ---- mask pad columns
  {
    int padc = NKVPAD - NKV;
    for (int idx = tid; idx < padc * 16; idx += 128) {
      int r = idx / padc, c = NKV + idx % padc;
      sc[(size_t)r * NKVPAD + c] = -1e30f;
    }
  }
  __syncthreads();

  // ---- phase 2: row max then sum(exp)
  {
    int r = tid >> 3, sub = tid & 7;
    float m = -1e30f;
    for (int c = sub; c < NKVPAD; c += 8)
      m = fmaxf(m, sc[(size_t)r * NKVPAD + c]);
    partial[r * 8 + sub] = m;
  }
  __syncthreads();
  if (tid < 16) {
    float m = partial[tid * 8];
#pragma unroll
    for (int i = 1; i < 8; i++) m = fmaxf(m, partial[tid * 8 + i]);
    rowmax[tid] = m;
  }
  __syncthreads();
  {
    int r = tid >> 3, sub = tid & 7;
    float m = rowmax[r], s = 0.f;
    for (int c = sub; c < NKVPAD; c += 8)
      s += __expf(sc[(size_t)r * NKVPAD + c] - m);
    partial[r * 8 + sub] = s;
  }
  __syncthreads();
  if (tid < 16) {
    float s = 0.f;
#pragma unroll
    for (int i = 0; i < 8; i++) s += partial[tid * 8 + i];
    rowinv[tid] = 1.f / s;
  }
  __syncthreads();

  // ---- phase 3: O = quant(P) * V ; wave owns 16-col slice of DH
  const float pm = rowmax[lmod], pinv = rowinv[lmod];
  const _Float16* vrow =
      Vt + (((size_t)b * Hh + h) * DHEAD + wave * 16 + lmod) * NKVPAD;
  v8f oacc = {};
  for (int ks = 0; ks < NKVPAD; ks += 32) {
    v16h pf = {};
#pragma unroll
    for (int j = 0; j < 16; j++) {
      int k = ks + j + lhalf * 16;  // A-frag K map
      float p = __expf(sc[(size_t)lmod * NKVPAD + k] - pm) * pinv;
      float q = fminf(rintf(p * rdw), 255.f) * dw;  // zp=0 weight quantizer
      pf[j] = (_Float16)q;
    }
    v16h vf = load_fragB(vrow + ks + lhalf * 16);
    oacc = wmma_f32_16x16x32(pf, vf, oacc);
  }
  const int colo = h * DHEAD + wave * 16 + lmod;
#pragma unroll
  for (int e = 0; e < 8; e++) {
    size_t row = (size_t)b * NQ + qt * 16 + lhalf * 8 + e;
    O[row * (HEADS * DHEAD) + colo] = (_Float16)oacc[e];
  }
}

// ---------------------------------------------------------------------------
// GEGLU: h f32 [M,8192] -> out f16 [M,4096]; exact gelu via erf
// ---------------------------------------------------------------------------
__global__ void geglu_kernel(const float* __restrict__ hbuf,
                             _Float16* __restrict__ out, long total) {
  long i = (long)blockIdx.x * 256 + threadIdx.x;
  if (i >= total) return;
  long row = i >> 12, col = i & 4095;
  float a = hbuf[row * 8192 + col];
  float g = hbuf[row * 8192 + 4096 + col];
  float gel = 0.5f * g * (1.f + erff(g * 0.70710678118654752f));
  out[i] = (_Float16)(a * gel);
}

// ---------------------------------------------------------------------------
// Host orchestration
// ---------------------------------------------------------------------------
extern "C" void kernel_launch(void* const* d_in, const int* in_sizes, int n_in,
                              void* d_out, int out_size, void* d_ws,
                              size_t ws_size, hipStream_t stream) {
  (void)in_sizes; (void)n_in; (void)out_size; (void)ws_size;
  const int Bb = 4, Nn = 1024, Dd = 1024, CTXN = 77, CTXD = 768;
  const int Mx = Bb * Nn;     // 4096
  const int Mc = Bb * CTXN;   // 308
  const int NKV2 = 96;        // 77 padded to a multiple of 32
  const float SCALE = 0.125f; // 64^-0.5

  const float* x    = (const float*)d_in[0];
  const float* ctx  = (const float*)d_in[1];
  const float* n1w  = (const float*)d_in[2];
  const float* n1b  = (const float*)d_in[3];
  const float* n2w  = (const float*)d_in[4];
  const float* n2b  = (const float*)d_in[5];
  const float* n3w  = (const float*)d_in[6];
  const float* n3b  = (const float*)d_in[7];
  const float* a1wq = (const float*)d_in[8];
  const float* a1wk = (const float*)d_in[9];
  const float* a1wv = (const float*)d_in[10];
  const float* a1wo = (const float*)d_in[11];
  const float* a1bo = (const float*)d_in[12];
  const float* a2wq = (const float*)d_in[13];
  const float* a2wk = (const float*)d_in[14];
  const float* a2wv = (const float*)d_in[15];
  const float* a2wo = (const float*)d_in[16];
  const float* a2bo = (const float*)d_in[17];
  const float* ffw1 = (const float*)d_in[18];
  const float* ffb1 = (const float*)d_in[19];
  const float* ffw2 = (const float*)d_in[20];
  const float* ffb2 = (const float*)d_in[21];
  const float* d1q = (const float*)d_in[22];
  const float* d1k = (const float*)d_in[23];
  const float* d1v = (const float*)d_in[24];
  const float* d1w = (const float*)d_in[25];
  const float* d2q = (const float*)d_in[26];
  const float* d2k = (const float*)d_in[27];
  const float* d2v = (const float*)d_in[28];
  const float* d2w = (const float*)d_in[29];

  size_t off = 0;
  auto alloc = [&](size_t bytes) -> char* {
    char* p = (char*)d_ws + off;
    off += (bytes + 255) & ~(size_t)255;
    return p;
  };
  _Float16* wq1b = (_Float16*)alloc((size_t)Dd * Dd * 2);
  _Float16* wk1b = (_Float16*)alloc((size_t)Dd * Dd * 2);
  _Float16* wv1b = (_Float16*)alloc((size_t)Dd * Dd * 2);
  _Float16* wo1b = (_Float16*)alloc((size_t)Dd * Dd * 2);
  _Float16* wq2b = (_Float16*)alloc((size_t)Dd * Dd * 2);
  _Float16* wk2b = (_Float16*)alloc((size_t)CTXD * Dd * 2);
  _Float16* wv2b = (_Float16*)alloc((size_t)CTXD * Dd * 2);
  _Float16* wo2b = (_Float16*)alloc((size_t)Dd * Dd * 2);
  _Float16* w1b  = (_Float16*)alloc((size_t)Dd * 8192 * 2);
  _Float16* w2b  = (_Float16*)alloc((size_t)4096 * Dd * 2);
  _Float16* ctxb = (_Float16*)alloc((size_t)Mc * CTXD * 2);
  _Float16* xln  = (_Float16*)alloc((size_t)Mx * Dd * 2);
  _Float16* qq   = (_Float16*)alloc((size_t)Mx * Dd * 2);   // [B,H,1024,64]
  _Float16* kk   = (_Float16*)alloc((size_t)Mx * Dd * 2);   // [B,H,npad,64]
  _Float16* vT   = (_Float16*)alloc((size_t)Mx * Dd * 2);   // [B,H,64,npad]
  _Float16* aout = (_Float16*)alloc((size_t)Mx * Dd * 2);
  float* x1 = (float*)alloc((size_t)Mx * Dd * 4);
  float* x2 = (float*)alloc((size_t)Mx * Dd * 4);
  float* qf = (float*)alloc((size_t)Mx * Dd * 4);
  float* kf = (float*)alloc((size_t)Mx * Dd * 4);
  float* vf = (float*)alloc((size_t)Mx * Dd * 4);
  float* hb = (float*)alloc((size_t)Mx * 8192 * 4);
  _Float16* gb = (_Float16*)alloc((size_t)Mx * 4096 * 2);

  auto cast = [&](const float* s, _Float16* d, long n) {
    int blocks = (int)((n + 255) / 256);
    if (blocks > 8192) blocks = 8192;
    cast_f16_kernel<<<blocks, 256, 0, stream>>>(s, d, n);
  };
  cast(a1wq, wq1b, (long)Dd * Dd);
  cast(a1wk, wk1b, (long)Dd * Dd);
  cast(a1wv, wv1b, (long)Dd * Dd);
  cast(a1wo, wo1b, (long)Dd * Dd);
  cast(a2wq, wq2b, (long)Dd * Dd);
  cast(a2wk, wk2b, (long)CTXD * Dd);
  cast(a2wv, wv2b, (long)CTXD * Dd);
  cast(a2wo, wo2b, (long)Dd * Dd);
  cast(ffw1, w1b, (long)Dd * 8192);
  cast(ffw2, w2b, (long)4096 * Dd);
  cast(ctx, ctxb, (long)Mc * CTXD);

  dim3 gP((Dd + BN - 1) / BN, (Mx + BM - 1) / BM);
  dim3 gC((Dd + BN - 1) / BN, (Mc + BM - 1) / BM);
  dim3 gF((8192 + BN - 1) / BN, (Mx + BM - 1) / BM);
  dim3 gQ1(1024, HEADS, Bb);
  dim3 gQ2(NKV2, HEADS, Bb);
  dim3 gA(Nn / 16, HEADS, Bb);

  // ===================== stage 1: quantized self-attention =================
  ln_f16_kernel<<<Mx, 256, 0, stream>>>(x, n1w, n1b, xln, Dd);
  gemm_f16_kernel<<<gP, 256, 0, stream>>>(xln, wq1b, qf, nullptr, nullptr, Mx, Dd, Dd);
  gemm_f16_kernel<<<gP, 256, 0, stream>>>(xln, wk1b, kf, nullptr, nullptr, Mx, Dd, Dd);
  gemm_f16_kernel<<<gP, 256, 0, stream>>>(xln, wv1b, vf, nullptr, nullptr, Mx, Dd, Dd);
  quant_heads_kernel<<<gQ1, DHEAD, 0, stream>>>(qf, qq, d1q, Nn, Nn);
  quant_heads_kernel<<<gQ1, DHEAD, 0, stream>>>(kf, kk, d1k, Nn, Nn);
  quant_headsT_kernel<<<gQ1, DHEAD, 0, stream>>>(vf, vT, d1v, Nn, Nn);
  attn_kernel<<<gA, 128, (size_t)16 * Nn * 4, stream>>>(qq, kk, vT, aout, d1w,
                                                        Nn, Nn, Nn, SCALE);
  gemm_f16_kernel<<<gP, 256, 0, stream>>>(aout, wo1b, x1, a1bo, x, Mx, Dd, Dd);

  // ===================== stage 2: quantized cross-attention ================
  ln_f16_kernel<<<Mx, 256, 0, stream>>>(x1, n2w, n2b, xln, Dd);
  gemm_f16_kernel<<<gP, 256, 0, stream>>>(xln, wq2b, qf, nullptr, nullptr, Mx, Dd, Dd);
  gemm_f16_kernel<<<gC, 256, 0, stream>>>(ctxb, wk2b, kf, nullptr, nullptr, Mc, Dd, CTXD);
  gemm_f16_kernel<<<gC, 256, 0, stream>>>(ctxb, wv2b, vf, nullptr, nullptr, Mc, Dd, CTXD);
  quant_heads_kernel<<<gQ1, DHEAD, 0, stream>>>(qf, qq, d2q, Nn, Nn);
  quant_heads_kernel<<<gQ2, DHEAD, 0, stream>>>(kf, kk, d2k, CTXN, NKV2);
  quant_headsT_kernel<<<gQ2, DHEAD, 0, stream>>>(vf, vT, d2v, CTXN, NKV2);
  attn_kernel<<<gA, 128, (size_t)16 * NKV2 * 4, stream>>>(qq, kk, vT, aout, d2w,
                                                          Nn, CTXN, NKV2, SCALE);
  gemm_f16_kernel<<<gP, 256, 0, stream>>>(aout, wo2b, x2, a2bo, x1, Mx, Dd, Dd);

  // ===================== stage 3: GEGLU feed-forward =======================
  ln_f16_kernel<<<Mx, 256, 0, stream>>>(x2, n3w, n3b, xln, Dd);
  gemm_f16_kernel<<<gF, 256, 0, stream>>>(xln, w1b, hb, ffb1, nullptr, Mx, 8192, Dd);
  {
    long tot = (long)Mx * 4096;
    geglu_kernel<<<(int)((tot + 255) / 256), 256, 0, stream>>>(hb, gb, tot);
  }
  gemm_f16_kernel<<<gP, 256, 0, stream>>>(gb, w2b, (float*)d_out, ffb2, x2, Mx,
                                          Dd, 4096);
}
